// GPS_MLP_Actor_30528627540347
// MI455X (gfx1250) — matrix operand order repaired and checked by reading, not compiled, not run on hardware
//
#include <hip/hip_runtime.h>
#include <hip/hip_bf16.h>
#include <math.h>

typedef __attribute__((ext_vector_type(16))) __bf16 v16bf;
typedef __attribute__((ext_vector_type(8)))  __bf16 v8bf;
typedef __attribute__((ext_vector_type(8)))  float  v8f;

#define N_NODES 4096
#define NEDGE   65536
#define IN_DIM  128
#define HID     256

// ---------------------------------------------------------------------------
// Generic WMMA GEMM:  C[M,N] = A[M,K] @ op(B)  (+bias)(+R1)(+R2)(relu)
//   transB=1 : B is [N,K] row-major (weight layout), C = A @ B^T
//   transB=0 : B is [K,N] row-major,                 C = A @ B
// Requirements (all call sites satisfy them): M % 128 == 0, K % 32 == 0.
// Only N is bounds-checked (N=64 for the per-head P@V GEMM).
// Residuals R1/R2 are indexed with ldc (same layout as C).
// Block tile 128x128, 8 waves (wave32), each wave 64x32 = 4x2 wmma tiles.
// Double-buffered LDS; global loads for tile k+1 issued before the WMMAs of
// tile k so load latency overlaps matrix work; one barrier per K-step.
// ---------------------------------------------------------------------------
#define BM 128
#define BN 128
#define BK 32
#define LDSS 40   // bf16 row stride in LDS: 32 + 8 pad (keeps 16B alignment)

__global__ __launch_bounds__(256)
void gemm_bf16_wmma(const float* __restrict__ A, int lda,
                    const float* __restrict__ B, int ldb, int transB,
                    float* __restrict__ C, int ldc,
                    const float* __restrict__ bias,
                    const float* __restrict__ R1,
                    const float* __restrict__ R2,
                    int relu, int M, int N, int K)
{
    __shared__ __align__(16) __bf16 As[2][BM * LDSS];
    __shared__ __align__(16) __bf16 Bs[2][BN * LDSS];

    const int tid  = threadIdx.x;
    const int lane = tid & 31;
    const int wave = tid >> 5;
    const int m16  = lane & 15;   // N-selector within 16x16 tile
    const int hi   = lane >> 4;   // lane-half selector

    const int row0 = blockIdx.y * BM;
    const int col0 = blockIdx.x * BN;
    const int wm0  = (wave >> 2) * 64;  // 2 wave rows
    const int wn0  = (wave & 3) * 32;   // 4 wave cols

    // staging coordinates (A and transB share the pattern)
    const int rS = tid >> 3;          // row 0..31 (+32 per it)
    const int kS = (tid & 7) << 2;    // k  0,4,..,28
    const int kT = tid >> 5;          // k 0..7 (+8 per it), non-trans B
    const int nT = (tid & 31) << 2;   // n 0..124, non-trans B

    v8f acc[4][2];
    #pragma unroll
    for (int i = 0; i < 4; ++i)
        #pragma unroll
        for (int j = 0; j < 2; ++j) {
            v8f z = {0.f,0.f,0.f,0.f,0.f,0.f,0.f,0.f};
            acc[i][j] = z;
        }

    float4 rA[4], rB[4];

    auto fetch = [&](int k0) {
        #pragma unroll
        for (int it = 0; it < 4; ++it) {
            int r = rS + it * 32;
            rA[it] = *(const float4*)(A + (size_t)(row0 + r) * lda + k0 + kS);
        }
        if (transB) {
            #pragma unroll
            for (int it = 0; it < 4; ++it) {
                int r = rS + it * 32;
                float4 v = make_float4(0.f, 0.f, 0.f, 0.f);
                if (col0 + r < N)
                    v = *(const float4*)(B + (size_t)(col0 + r) * ldb + k0 + kS);
                rB[it] = v;
            }
        } else {
            #pragma unroll
            for (int it = 0; it < 4; ++it) {
                int kk = kT + it * 8;
                const float* s = B + (size_t)(k0 + kk) * ldb + col0 + nT;
                float4 v = make_float4(0.f, 0.f, 0.f, 0.f);
                if (col0 + nT + 3 < N) {
                    v = *(const float4*)s;
                } else {
                    if (col0 + nT     < N) v.x = s[0];
                    if (col0 + nT + 1 < N) v.y = s[1];
                    if (col0 + nT + 2 < N) v.z = s[2];
                }
                rB[it] = v;
            }
        }
    };

    auto stash = [&](int buf) {
        #pragma unroll
        for (int it = 0; it < 4; ++it) {
            int r = rS + it * 32;
            __bf16* d = &As[buf][r * LDSS + kS];
            d[0] = (__bf16)rA[it].x; d[1] = (__bf16)rA[it].y;
            d[2] = (__bf16)rA[it].z; d[3] = (__bf16)rA[it].w;
        }
        if (transB) {
            #pragma unroll
            for (int it = 0; it < 4; ++it) {
                int r = rS + it * 32;
                __bf16* d = &Bs[buf][r * LDSS + kS];
                d[0] = (__bf16)rB[it].x; d[1] = (__bf16)rB[it].y;
                d[2] = (__bf16)rB[it].z; d[3] = (__bf16)rB[it].w;
            }
        } else {
            #pragma unroll
            for (int it = 0; it < 4; ++it) {
                int kk = kT + it * 8;
                Bs[buf][(nT + 0) * LDSS + kk] = (__bf16)rB[it].x;
                Bs[buf][(nT + 1) * LDSS + kk] = (__bf16)rB[it].y;
                Bs[buf][(nT + 2) * LDSS + kk] = (__bf16)rB[it].z;
                Bs[buf][(nT + 3) * LDSS + kk] = (__bf16)rB[it].w;
            }
        }
    };

    const int nk = K / BK;            // K % 32 == 0 guaranteed
    fetch(0);
    stash(0);

    for (int kb = 0; kb < nk; ++kb) {
        const int cur = kb & 1;
        __syncthreads();              // stores to buf[cur] visible; buf[cur^1] free

        if (kb + 1 < nk) fetch((kb + 1) * BK);   // overlap with WMMAs below
        if (kb + 2 < nk) {                       // warm L2 two tiles ahead
            int pk = (kb + 2) * BK;
            __builtin_prefetch(A + (size_t)(row0 + rS) * lda + pk + kS, 0, 1);
            if (transB) {
                if (col0 + rS < N)
                    __builtin_prefetch(B + (size_t)(col0 + rS) * ldb + pk + kS, 0, 1);
            } else {
                __builtin_prefetch(B + (size_t)(pk + kT) * ldb + col0 + nT, 0, 1);
            }
        }

        // ---- build fragments per ISA 7.12.2 layouts and issue WMMA
        v16bf afrag[4];
        #pragma unroll
        for (int mt = 0; mt < 4; ++mt) {
            const __bf16* base = &As[cur][(wm0 + mt * 16 + m16) * LDSS];
            v8bf lo = *(const v8bf*)(base + hi * 8);        // K = hi*8 .. +7
            v8bf hv = *(const v8bf*)(base + 16 + hi * 8);   // K = 16+hi*8 .. +7
            afrag[mt] = __builtin_shufflevector(lo, hv,
                0,1,2,3,4,5,6,7,8,9,10,11,12,13,14,15);
        }
        v16bf bfrag[2];
        #pragma unroll
        for (int nt = 0; nt < 2; ++nt) {
            const __bf16* base = &Bs[cur][(wn0 + nt * 16 + m16) * LDSS];
            v8bf lo = *(const v8bf*)(base + hi * 8);        // K = hi*8 .. +7
            v8bf hv = *(const v8bf*)(base + 16 + hi * 8);   // K = 16+hi*8 .. +7
            bfrag[nt] = __builtin_shufflevector(lo, hv,
                0,1,2,3,4,5,6,7,8,9,10,11,12,13,14,15);
        }
        #pragma unroll
        for (int mt = 0; mt < 4; ++mt)
            #pragma unroll
            for (int nt = 0; nt < 2; ++nt)
                acc[mt][nt] = __builtin_amdgcn_wmma_f32_16x16x32_bf16(
                    false, afrag[mt], false, bfrag[nt],
                    (short)0, acc[mt][nt], false, false);

        if (kb + 1 < nk) stash(cur ^ 1);
    }

    // ---- epilogue: C layout lane -> (M = j + 8*hi, N = m16) within 16x16 tile
    #pragma unroll
    for (int mt = 0; mt < 4; ++mt) {
        #pragma unroll
        for (int nt = 0; nt < 2; ++nt) {
            int c = col0 + wn0 + nt * 16 + m16;
            if (c >= N) continue;
            float bv = bias ? bias[c] : 0.0f;
            #pragma unroll
            for (int j = 0; j < 8; ++j) {
                int r = row0 + wm0 + mt * 16 + hi * 8 + j;
                size_t off = (size_t)r * ldc + c;
                float v = acc[mt][nt][j] + bv;
                if (R1) v += R1[off];
                if (R2) v += R2[off];
                if (relu) v = fmaxf(v, 0.0f);
                C[off] = v;
            }
        }
    }
}

// ---------------------------------------------------------------------------
// Small helper kernels
// ---------------------------------------------------------------------------
__global__ void fill_f32(float* p, float v, int n) {
    int i = blockIdx.x * blockDim.x + threadIdx.x;
    if (i < n) p[i] = v;
}

__global__ void deg_count(const int* __restrict__ dst, float* deg, int E) {
    int e = blockIdx.x * blockDim.x + threadIdx.x;
    if (e < E) atomicAdd(&deg[dst[e]], 1.0f);
}

__global__ void rsqrt_inplace(float* p, int n) {
    int i = blockIdx.x * blockDim.x + threadIdx.x;
    if (i < n) p[i] = rsqrtf(p[i]);
}

// out = xlin * dinv^2 (self-loop) + gcn_bias + residual
__global__ void gcn_self(float* __restrict__ out, const float* __restrict__ xlin,
                         const float* __restrict__ dinv, const float* __restrict__ b,
                         const float* __restrict__ resid, int n, int h) {
    int i = blockIdx.x * blockDim.x + threadIdx.x;
    if (i < n * h) {
        int node = i / h, c = i - node * h;
        float di = dinv[node];
        out[i] = xlin[i] * di * di + b[c] + resid[i];
    }
}

// one block per edge, one thread per channel (H == blockDim.x == 256)
__global__ void edge_agg(const int* __restrict__ src, const int* __restrict__ dst,
                         const float* __restrict__ dinv, const float* __restrict__ xlin,
                         float* __restrict__ out, int H) {
    int e = blockIdx.x;
    int s = src[e], d = dst[e];
    float coef = dinv[s] * dinv[d];
    int c = threadIdx.x;
    atomicAdd(&out[(size_t)d * H + c], xlin[(size_t)s * H + c] * coef);
}

// row-wise softmax of S (scale applied pre-exp), in place; one block per row
__global__ __launch_bounds__(256)
void softmax_rows(float* __restrict__ S, int cols, float scale) {
    __shared__ float red[256];
    float* p = S + (size_t)blockIdx.x * cols;
    int t = threadIdx.x;
    float m = -INFINITY;
    for (int c = t; c < cols; c += 256) m = fmaxf(m, p[c] * scale);
    red[t] = m; __syncthreads();
    for (int s = 128; s > 0; s >>= 1) {
        if (t < s) red[t] = fmaxf(red[t], red[t + s]);
        __syncthreads();
    }
    m = red[0]; __syncthreads();
    float sum = 0.0f;
    for (int c = t; c < cols; c += 256) {
        float e = __expf(p[c] * scale - m);
        p[c] = e; sum += e;
    }
    red[t] = sum; __syncthreads();
    for (int s = 128; s > 0; s >>= 1) {
        if (t < s) red[t] += red[t + s];
        __syncthreads();
    }
    float inv = 1.0f / red[0];
    for (int c = t; c < cols; c += 256) p[c] *= inv;
}

// D = (D + D^T) / 2, in place; each (i<j) pair handled by exactly one thread
__global__ void symmetrize(float* __restrict__ D, int n) {
    int j = blockIdx.x * 16 + threadIdx.x;
    int i = blockIdx.y * 16 + threadIdx.y;
    if (i < n && j < n && j > i) {
        size_t a = (size_t)i * n + j, b = (size_t)j * n + i;
        float m = 0.5f * (D[a] + D[b]);
        D[a] = m; D[b] = m;
    }
}

// ---------------------------------------------------------------------------
// Driver
// ---------------------------------------------------------------------------
extern "C" void kernel_launch(void* const* d_in, const int* in_sizes, int n_in,
                              void* d_out, int out_size, void* d_ws, size_t ws_size,
                              hipStream_t stream)
{
    const float* x     = (const float*)d_in[0];
    const int*   ei    = (const int*)d_in[1];
    const float* pre_w = (const float*)d_in[2];
    const float* pre_b = (const float*)d_in[3];
    const float* W[2][10];
    for (int l = 0; l < 2; ++l)
        for (int k = 0; k < 10; ++k)
            W[l][k] = (const float*)d_in[4 + l * 10 + k];
    const float* mlp_w1 = (const float*)d_in[24];
    const float* mlp_b1 = (const float*)d_in[25];
    const float* mlp_w2 = (const float*)d_in[26];
    const float* mlp_b2 = (const float*)d_in[27];

    const int* src = ei;
    const int* dst = ei + NEDGE;

    float* ws = (float*)d_ws;
    const size_t NH = (size_t)N_NODES * HID;
    float* h   = ws;                    // [N,H]   current activations
    float* t1  = ws + NH;               // [N,H]   gcn_lin / attn concat / mlp mid
    float* t2  = ws + 2 * NH;           // [N,H]   h1 (GCN branch)
    float* qkv = ws + 3 * NH;           // [N,3H]  qkv; later reused as 'out'
    float* ffm = ws + 6 * NH;           // [N,2H]  FFN hidden
    float* deg = ws + 8 * NH;           // [N]     degree -> dinv
    float* S   = ws + 8 * NH + N_NODES; // [N,N]   per-head attention scores

    auto gemm = [&](const float* A, int lda, const float* B, int ldb, int tB,
                    float* Cc, int ldc, const float* bias, const float* R1,
                    const float* R2, int relu, int M, int Nn, int K) {
        dim3 g((Nn + BN - 1) / BN, (M + BM - 1) / BM);
        gemm_bf16_wmma<<<g, 256, 0, stream>>>(A, lda, B, ldb, tB, Cc, ldc,
                                              bias, R1, R2, relu, M, Nn, K);
    };

    // h = relu(x @ pre_w^T + pre_b)
    gemm(x, IN_DIM, pre_w, IN_DIM, 1, h, HID, pre_b, nullptr, nullptr, 1,
         N_NODES, HID, IN_DIM);

    for (int l = 0; l < 2; ++l) {
        // ---- GCN branch: t2 = h1 = GCN(h) + gcn_b + h
        gemm(h, HID, W[l][0], HID, 1, t1, HID, nullptr, nullptr, nullptr, 0,
             N_NODES, HID, HID);
        fill_f32<<<(N_NODES + 255) / 256, 256, 0, stream>>>(deg, 1.0f, N_NODES);
        deg_count<<<(NEDGE + 255) / 256, 256, 0, stream>>>(dst, deg, NEDGE);
        rsqrt_inplace<<<(N_NODES + 255) / 256, 256, 0, stream>>>(deg, N_NODES);
        gcn_self<<<((int)(NH) + 255) / 256, 256, 0, stream>>>(t2, t1, deg,
                                                              W[l][1], h, N_NODES, HID);
        edge_agg<<<NEDGE, HID, 0, stream>>>(src, dst, deg, t1, t2, HID);

        // ---- Attention branch
        gemm(h, HID, W[l][2], HID, 1, qkv, 3 * HID, W[l][3], nullptr, nullptr, 0,
             N_NODES, 3 * HID, HID);
        for (int hd = 0; hd < 4; ++hd) {
            const float* Q  = qkv + hd * 64;
            const float* Kk = qkv + HID + hd * 64;
            const float* V  = qkv + 2 * HID + hd * 64;
            // S = Q @ K^T   (K stored [N,64] row-major -> transB path)
            gemm(Q, 3 * HID, Kk, 3 * HID, 1, S, N_NODES, nullptr, nullptr,
                 nullptr, 0, N_NODES, N_NODES, 64);
            softmax_rows<<<N_NODES, 256, 0, stream>>>(S, N_NODES, 0.125f);
            // attn[:, hd*64:+64] = P @ V   (non-transposed B)
            gemm(S, N_NODES, V, 3 * HID, 0, t1 + hd * 64, HID, nullptr, nullptr,
                 nullptr, 0, N_NODES, 64, N_NODES);
        }
        // ---- out = h2 + h1 = (attn @ out_w^T + out_b + h) + t2   (qkv dead -> reuse)
        float* outb = qkv;
        gemm(t1, HID, W[l][4], HID, 1, outb, HID, W[l][5], h, t2, 0,
             N_NODES, HID, HID);
        // ---- FFN + residual + outer relu -> new h
        gemm(outb, HID, W[l][6], HID, 1, ffm, 2 * HID, W[l][7], nullptr, nullptr, 1,
             N_NODES, 2 * HID, HID);
        gemm(ffm, 2 * HID, W[l][8], 2 * HID, 1, h, HID, W[l][9], outb, nullptr, 1,
             N_NODES, HID, 2 * HID);
    }

    // ---- output head
    gemm(h, HID, mlp_w1, HID, 1, t1, HID, mlp_b1, nullptr, nullptr, 1,
         N_NODES, HID, HID);
    float* dout = (float*)d_out;
    gemm(t1, HID, mlp_w2, HID, 1, dout, N_NODES, mlp_b2, nullptr, nullptr, 0,
         N_NODES, N_NODES, HID);
    symmetrize<<<dim3(N_NODES / 16, N_NODES / 16), dim3(16, 16), 0, stream>>>(
        dout, N_NODES);
}
